// SwinTransformerBlock3D_14783277433306
// MI455X (gfx1250) — matrix operand order repaired and checked by reading, not compiled, not run on hardware
//
#include <hip/hip_runtime.h>
#include <cmath>

typedef __attribute__((ext_vector_type(16))) _Float16 v16h;
typedef __attribute__((ext_vector_type(8)))  _Float16 v8h;
typedef __attribute__((ext_vector_type(4)))  _Float16 v4h;
typedef __attribute__((ext_vector_type(8)))  float    v8f;
typedef __attribute__((ext_vector_type(4)))  float    v4f;

#define NWIN   512      // total windows (B=4, 2x8x8 per image)
#define NPAD   128      // padded tokens per window
#define NTOK   125      // valid tokens per window
#define CDIM   128
#define NTOKENS 64000   // 4*10*40*40
#define QSCALE 0.17677669529663687f  // 32^-0.5

// ---------- WMMA fragment helpers (CDNA5 layouts, wave32) ----------
static __device__ __forceinline__ v16h hcat(v8h a, v8h b) {
  v16h r;
#pragma unroll
  for (int i = 0; i < 8; ++i) { r[i] = a[i]; r[i + 8] = b[i]; }
  return r;
}
// A fragment 16x32 f16: lane&15 = row M; lanes 0-15 hold K[0..7]+K[16..23],
// lanes 16-31 hold K[8..15]+K[24..31]. Two contiguous 16B loads per lane.
static __device__ __forceinline__ v16h loadA(const _Float16* base, int ld, int kc) {
  const int lane = threadIdx.x & 31;
  const _Float16* p = base + (size_t)(lane & 15) * ld + kc * 32 + (lane >> 4) * 8;
  return hcat(*(const v8h*)p, *(const v8h*)(p + 16));
}
// B fragment 32x16 f16: lane&15 = column N; lanes 0-15 hold K[0..7]+K[16..23],
// lanes 16-31 hold K[8..15]+K[24..31]. "base" is row-major over the N dimension
// (i.e. B[k][n] = base[n*ld + k]) -> two contiguous 16B loads per lane.
static __device__ __forceinline__ v16h loadB(const _Float16* base, int ld, int kc) {
  const int lane = threadIdx.x & 31;
  const _Float16* p = base + (size_t)(lane & 15) * ld + kc * 32 + (lane >> 4) * 8;
  return hcat(*(const v8h*)p, *(const v8h*)(p + 16));
}
static __device__ __forceinline__ v8f wmma16(v16h a, v16h b, v8f c) {
  return __builtin_amdgcn_wmma_f32_16x16x32_f16(false, a, false, b, (short)0, c, false, false);
}
static __device__ __forceinline__ v8f splat8(float v) {
  v8f r;
#pragma unroll
  for (int i = 0; i < 8; ++i) r[i] = v;
  return r;
}

// ---------- K0: fp32 -> f16 convert ----------
__global__ void k_cvt(const float* __restrict__ s, _Float16* __restrict__ d, int n) {
  int i = blockIdx.x * blockDim.x + threadIdx.x;
  if (i < n) d[i] = (_Float16)s[i];
}

// ---------- K1: LN1 + roll(-2) + window partition (padded, f16) ----------
__global__ void k_ln1(const float* __restrict__ x, const float* __restrict__ g,
                      const float* __restrict__ bt, _Float16* __restrict__ xw) {
  const int wid = blockIdx.x * 8 + (threadIdx.x >> 5);  // (win,row)
  const int lane = threadIdx.x & 31;
  const int win = wid >> 7, row = wid & 127;
  _Float16* dst = xw + (size_t)wid * CDIM;
  if (row >= NTOK) {
    v4h z; z[0] = (_Float16)0.f; z[1] = (_Float16)0.f; z[2] = (_Float16)0.f; z[3] = (_Float16)0.f;
    *(v4h*)(dst + lane * 4) = z;
    return;
  }
  const int b  = win >> 7, dw = (win >> 6) & 1, hw = (win >> 3) & 7, ww = win & 7;
  const int zd = row / 25, zh = (row % 25) / 5, zw = row % 5;
  const int ds = (dw * 5 + zd + 2) % 10;
  const int hs = (hw * 5 + zh + 2) % 40;
  const int wsrc = (ww * 5 + zw + 2) % 40;
  const size_t t = (((size_t)b * 10 + ds) * 40 + hs) * 40 + wsrc;
  v4f xv = *(const v4f*)(x + t * CDIM + lane * 4);
  float s  = xv[0] + xv[1] + xv[2] + xv[3];
  float ss = xv[0]*xv[0] + xv[1]*xv[1] + xv[2]*xv[2] + xv[3]*xv[3];
#pragma unroll
  for (int m = 16; m >= 1; m >>= 1) { s += __shfl_xor(s, m, 32); ss += __shfl_xor(ss, m, 32); }
  const float mean = s * (1.f / 128.f);
  const float var  = ss * (1.f / 128.f) - mean * mean;
  const float inv  = rsqrtf(var + 1e-5f);
  v4h o;
#pragma unroll
  for (int i = 0; i < 4; ++i) {
    const int c = lane * 4 + i;
    o[i] = (_Float16)((xv[i] - mean) * inv * g[c] + bt[c]);
  }
  *(v4h*)(dst + lane * 4) = o;
}

// ---------- K2: QKV GEMM  out[win] = xw[win] @ Wqkv^T + b (M=128,N=384,K=128) ----------
// Two N-tiles per iteration sharing the A fragments; q/k/v routing constants are
// derived from the (fully unrolled) tile index only -> branch-free epilogue.
__global__ void k_qkv(const _Float16* __restrict__ xw, const _Float16* __restrict__ qw,
                      const float* __restrict__ qb, _Float16* __restrict__ qh,
                      _Float16* __restrict__ kh, _Float16* __restrict__ vt) {
  const int win = blockIdx.x;
  const int wv = threadIdx.x >> 5, lane = threadIdx.x & 31;
  const int j = lane & 15, hi = lane >> 4;
  const _Float16* Ab = xw + ((size_t)win * NPAD + wv * 16) * CDIM;
  const v16h a0 = loadA(Ab, CDIM, 0), a1 = loadA(Ab, CDIM, 1),
             a2 = loadA(Ab, CDIM, 2), a3 = loadA(Ab, CDIM, 3);
#pragma unroll
  for (int np = 0; np < 12; ++np) {
    const int n0 = 2 * np, n1 = 2 * np + 1;
    const _Float16* B0 = qw + (size_t)(n0 * 16) * CDIM;
    const _Float16* B1 = B0 + 16 * CDIM;
    if (np < 11) __builtin_prefetch(B0 + 32 * CDIM + (size_t)lane * 128, 0, 1);
    v8f acc0 = splat8(qb[n0 * 16 + j]);
    v8f acc1 = splat8(qb[n1 * 16 + j]);
    acc0 = wmma16(a0, loadB(B0, CDIM, 0), acc0);
    acc1 = wmma16(a0, loadB(B1, CDIM, 0), acc1);
    acc0 = wmma16(a1, loadB(B0, CDIM, 1), acc0);
    acc1 = wmma16(a1, loadB(B1, CDIM, 1), acc1);
    acc0 = wmma16(a2, loadB(B0, CDIM, 2), acc0);
    acc1 = wmma16(a2, loadB(B1, CDIM, 2), acc1);
    acc0 = wmma16(a3, loadB(B0, CDIM, 3), acc0);
    acc1 = wmma16(a3, loadB(B1, CDIM, 3), acc1);
#pragma unroll
    for (int t = 0; t < 2; ++t) {
      const int nt = t ? n1 : n0;
      const v8f acc = t ? acc1 : acc0;
      const int which = nt >> 3;            // 0=q 1=k 2=v (compile-time)
      const int head  = (nt >> 1) & 3;      // compile-time
      const int d     = (nt & 1) * 16 + j;
      const size_t hb = (size_t)win * 4 + head;
#pragma unroll
      for (int r = 0; r < 8; ++r) {
        const int row = wv * 16 + r + 8 * hi;
        const float v = acc[r];
        if (which == 0)      qh[(hb * NPAD + row) * 32 + d] = (_Float16)(v * QSCALE);
        else if (which == 1) kh[(hb * NPAD + row) * 32 + d] = (_Float16)v;
        else                 vt[(hb * 32 + d) * NPAD + row] = (_Float16)v;  // transposed
      }
    }
  }
}

// ---------- K3: attention per (window, head): softmax(q k^T + bias + mask) v ----------
__global__ void k_attn(const _Float16* __restrict__ qh, const _Float16* __restrict__ kh,
                       const _Float16* __restrict__ vt, const float* __restrict__ relb,
                       _Float16* __restrict__ ao) {
  __shared__ _Float16 P[NPAD * NPAD];
  __shared__ int pk[NTOK];   // f(n)=zd*81+zh*9+zw in low 16 bits, region id in high bits
  const int win = blockIdx.x >> 2, head = blockIdx.x & 3;
  const int wv = threadIdx.x >> 5, lane = threadIdx.x & 31;
  const int j = lane & 15, hi = lane >> 4;

  if (threadIdx.x < NTOK) {
    const int n = threadIdx.x;
    const int zd = n / 25, zh = (n % 25) / 5, zw = n % 5;
    const int dw = (win >> 6) & 1, hw = (win >> 3) & 7, ww = win & 7;
    const int d = dw * 5 + zd, h = hw * 5 + zh, w = ww * 5 + zw;
    const int rd = d < 5 ? 0 : (d < 8 ? 1 : 2);
    const int rh = h < 35 ? 0 : (h < 38 ? 1 : 2);
    const int rw = w < 35 ? 0 : (w < 38 ? 1 : 2);
    pk[n] = (zd * 81 + zh * 9 + zw) | ((rd * 9 + rh * 3 + rw) << 16);
  }
  __syncthreads();

  const size_t hb = (size_t)win * 4 + head;
  const _Float16* qp = qh + hb * NPAD * 32;
  const _Float16* kp = kh + hb * NPAD * 32;
  const _Float16* vp = vt + hb * 32 * NPAD;

  const v16h aq = loadA(qp + (size_t)wv * 16 * 32, 32, 0);
  v8f s[8];
#pragma unroll
  for (int nt = 0; nt < 8; ++nt)
    s[nt] = wmma16(aq, loadB(kp + (size_t)nt * 16 * 32, 32, 0), splat8(0.f));

  int prow[8], pcol[8];
#pragma unroll
  for (int r = 0; r < 8; ++r) {
    const int row = wv * 16 + r + 8 * hi;
    prow[r] = (row < NTOK) ? pk[row] : -1;
  }
#pragma unroll
  for (int nt = 0; nt < 8; ++nt) {
    const int col = nt * 16 + j;
    pcol[nt] = (col < NTOK) ? pk[col] : -1;
  }
  // bias + shift-mask + padding
#pragma unroll
  for (int nt = 0; nt < 8; ++nt) {
#pragma unroll
    for (int r = 0; r < 8; ++r) {
      float v = s[nt][r];
      if (pcol[nt] < 0) v = -1e30f;
      else if (prow[r] >= 0) {
        const int idx = (prow[r] & 0xffff) - (pcol[nt] & 0xffff) + 364;
        v += relb[idx * 4 + head];
        if ((prow[r] >> 16) != (pcol[nt] >> 16)) v -= 100.f;
      }
      s[nt][r] = v;
    }
  }
  // softmax over 128 cols (per row = 8 regs x 16 lanes of one half-wave)
#pragma unroll
  for (int r = 0; r < 8; ++r) {
    float mx = -1e30f;
#pragma unroll
    for (int nt = 0; nt < 8; ++nt) mx = fmaxf(mx, s[nt][r]);
#pragma unroll
    for (int mm = 8; mm >= 1; mm >>= 1) mx = fmaxf(mx, __shfl_xor(mx, mm, 32));
    float sum = 0.f;
#pragma unroll
    for (int nt = 0; nt < 8; ++nt) { const float e = __expf(s[nt][r] - mx); s[nt][r] = e; sum += e; }
#pragma unroll
    for (int mm = 8; mm >= 1; mm >>= 1) sum += __shfl_xor(sum, mm, 32);
    const float inv = 1.f / sum;
    const int row = wv * 16 + r + 8 * hi;
#pragma unroll
    for (int nt = 0; nt < 8; ++nt) P[row * NPAD + nt * 16 + j] = (_Float16)(s[nt][r] * inv);
  }
  __syncthreads();

  // out = P @ v  (M=128, N=32, K=128); B from transposed v -> contiguous
  v8f o0 = splat8(0.f), o1 = splat8(0.f);
#pragma unroll
  for (int kc = 0; kc < 4; ++kc) {
    const v16h ap = loadA(&P[wv * 16 * NPAD], NPAD, kc);
    o0 = wmma16(ap, loadB(vp, NPAD, kc), o0);
    o1 = wmma16(ap, loadB(vp + 16 * NPAD, NPAD, kc), o1);
  }
#pragma unroll
  for (int r = 0; r < 8; ++r) {
    const int row = wv * 16 + r + 8 * hi;
    _Float16* dst = ao + ((size_t)win * NPAD + row) * CDIM + head * 32;
    dst[j]      = (_Float16)o0[r];
    dst[16 + j] = (_Float16)o1[r];
  }
}

// ---------- K4: proj GEMM + window reverse + roll(+2) + shortcut residual ----------
__global__ void k_proj(const _Float16* __restrict__ ao, const _Float16* __restrict__ pw,
                       const float* __restrict__ pb, const float* __restrict__ x,
                       float* __restrict__ y) {
  const int win = blockIdx.x;
  const int wv = threadIdx.x >> 5, lane = threadIdx.x & 31;
  const int j = lane & 15, hi = lane >> 4;
  const _Float16* Ab = ao + ((size_t)win * NPAD + wv * 16) * CDIM;
  const v16h a0 = loadA(Ab, CDIM, 0), a1 = loadA(Ab, CDIM, 1),
             a2 = loadA(Ab, CDIM, 2), a3 = loadA(Ab, CDIM, 3);
  const int b  = win >> 7, dw = (win >> 6) & 1, hw = (win >> 3) & 7, ww = win & 7;
#pragma unroll
  for (int np = 0; np < 4; ++np) {
    const int n0 = 2 * np, n1 = 2 * np + 1;
    const _Float16* B0 = pw + (size_t)(n0 * 16) * CDIM;
    const _Float16* B1 = B0 + 16 * CDIM;
    v8f acc0 = splat8(pb[n0 * 16 + j]);
    v8f acc1 = splat8(pb[n1 * 16 + j]);
    acc0 = wmma16(a0, loadB(B0, CDIM, 0), acc0);
    acc1 = wmma16(a0, loadB(B1, CDIM, 0), acc1);
    acc0 = wmma16(a1, loadB(B0, CDIM, 1), acc0);
    acc1 = wmma16(a1, loadB(B1, CDIM, 1), acc1);
    acc0 = wmma16(a2, loadB(B0, CDIM, 2), acc0);
    acc1 = wmma16(a2, loadB(B1, CDIM, 2), acc1);
    acc0 = wmma16(a3, loadB(B0, CDIM, 3), acc0);
    acc1 = wmma16(a3, loadB(B1, CDIM, 3), acc1);
#pragma unroll
    for (int t = 0; t < 2; ++t) {
      const int nt = t ? n1 : n0;
      const v8f acc = t ? acc1 : acc0;
#pragma unroll
      for (int r = 0; r < 8; ++r) {
        const int row = wv * 16 + r + 8 * hi;
        if (row < NTOK) {
          const int zd = row / 25, zh = (row % 25) / 5, zw = row % 5;
          const int df = (dw * 5 + zd + 2) % 10;
          const int hf = (hw * 5 + zh + 2) % 40;
          const int wf = (ww * 5 + zw + 2) % 40;
          const size_t tk = (((size_t)b * 10 + df) * 40 + hf) * 40 + wf;
          const int c = nt * 16 + j;
          y[tk * CDIM + c] = x[tk * CDIM + c] + acc[r];
        }
      }
    }
  }
}

// ---------- K5: LN2 (flat over tokens) ----------
__global__ void k_ln2(const float* __restrict__ y, const float* __restrict__ g,
                      const float* __restrict__ bt, _Float16* __restrict__ mh) {
  const int wid = blockIdx.x * 8 + (threadIdx.x >> 5);
  const int lane = threadIdx.x & 31;
  v4f xv = *(const v4f*)(y + (size_t)wid * CDIM + lane * 4);
  float s  = xv[0] + xv[1] + xv[2] + xv[3];
  float ss = xv[0]*xv[0] + xv[1]*xv[1] + xv[2]*xv[2] + xv[3]*xv[3];
#pragma unroll
  for (int m = 16; m >= 1; m >>= 1) { s += __shfl_xor(s, m, 32); ss += __shfl_xor(ss, m, 32); }
  const float mean = s * (1.f / 128.f);
  const float inv  = rsqrtf(ss * (1.f / 128.f) - mean * mean + 1e-5f);
  v4h o;
#pragma unroll
  for (int i = 0; i < 4; ++i) {
    const int c = lane * 4 + i;
    o[i] = (_Float16)((xv[i] - mean) * inv * g[c] + bt[c]);
  }
  *(v4h*)(mh + (size_t)wid * CDIM + lane * 4) = o;
}

// ---------- K6: MLP1 + exact GELU (M=64000, N=512, K=128) ----------
__global__ void k_mlp1(const _Float16* __restrict__ mh, const _Float16* __restrict__ w1,
                       const float* __restrict__ b1, _Float16* __restrict__ h1) {
  const int mt = blockIdx.x * 8 + (threadIdx.x >> 5);  // 0..3999
  const int lane = threadIdx.x & 31;
  const int j = lane & 15, hi = lane >> 4;
  const _Float16* Ab = mh + (size_t)mt * 16 * CDIM;
  const v16h a0 = loadA(Ab, CDIM, 0), a1 = loadA(Ab, CDIM, 1),
             a2 = loadA(Ab, CDIM, 2), a3 = loadA(Ab, CDIM, 3);
#pragma unroll 2
  for (int np = 0; np < 16; ++np) {
    const int n0 = 2 * np, n1 = 2 * np + 1;
    const _Float16* B0 = w1 + (size_t)(n0 * 16) * CDIM;
    const _Float16* B1 = B0 + 16 * CDIM;
    if (np < 15) __builtin_prefetch(B0 + 32 * CDIM + (size_t)lane * 128, 0, 1);
    v8f acc0 = splat8(b1[n0 * 16 + j]);
    v8f acc1 = splat8(b1[n1 * 16 + j]);
    acc0 = wmma16(a0, loadB(B0, CDIM, 0), acc0);
    acc1 = wmma16(a0, loadB(B1, CDIM, 0), acc1);
    acc0 = wmma16(a1, loadB(B0, CDIM, 1), acc0);
    acc1 = wmma16(a1, loadB(B1, CDIM, 1), acc1);
    acc0 = wmma16(a2, loadB(B0, CDIM, 2), acc0);
    acc1 = wmma16(a2, loadB(B1, CDIM, 2), acc1);
    acc0 = wmma16(a3, loadB(B0, CDIM, 3), acc0);
    acc1 = wmma16(a3, loadB(B1, CDIM, 3), acc1);
#pragma unroll
    for (int t = 0; t < 2; ++t) {
      const int nt = t ? n1 : n0;
      const v8f acc = t ? acc1 : acc0;
#pragma unroll
      for (int r = 0; r < 8; ++r) {
        const float v = acc[r];
        const float ge = 0.5f * v * (1.f + erff(v * 0.70710678118f));
        h1[((size_t)mt * 16 + r + 8 * hi) * 512 + nt * 16 + j] = (_Float16)ge;
      }
    }
  }
}

// ---------- K7: MLP2 + residual into d_out (M=64000, N=128, K=512) ----------
__global__ void k_mlp2(const _Float16* __restrict__ h1, const _Float16* __restrict__ w2,
                       const float* __restrict__ b2, float* __restrict__ out) {
  const int mt = blockIdx.x * 8 + (threadIdx.x >> 5);
  const int lane = threadIdx.x & 31;
  const int j = lane & 15, hi = lane >> 4;
  v8f acc[8];
#pragma unroll
  for (int nt = 0; nt < 8; ++nt) acc[nt] = splat8(b2[nt * 16 + j]);
  const _Float16* Ab = h1 + (size_t)mt * 16 * 512;
#pragma unroll 4
  for (int kc = 0; kc < 16; ++kc) {
    const v16h a = loadA(Ab, 512, kc);
#pragma unroll
    for (int nt = 0; nt < 8; ++nt)
      acc[nt] = wmma16(a, loadB(w2 + (size_t)(nt * 16) * 512, 512, kc), acc[nt]);
  }
#pragma unroll
  for (int nt = 0; nt < 8; ++nt) {
#pragma unroll
    for (int r = 0; r < 8; ++r) {
      const size_t row = (size_t)mt * 16 + r + 8 * hi;
      out[row * CDIM + nt * 16 + j] += acc[nt][r];
    }
  }
}

extern "C" void kernel_launch(void* const* d_in, const int* in_sizes, int n_in,
                              void* d_out, int out_size, void* d_ws, size_t ws_size,
                              hipStream_t stream) {
  (void)in_sizes; (void)n_in; (void)out_size; (void)ws_size;
  const float* x    = (const float*)d_in[0];
  // d_in[1] (shift mask) recomputed analytically in-kernel
  const float* n1g  = (const float*)d_in[2];
  const float* n1b  = (const float*)d_in[3];
  const float* qkvw = (const float*)d_in[4];
  const float* qkvb = (const float*)d_in[5];
  const float* relb = (const float*)d_in[6];
  const float* pjw  = (const float*)d_in[7];
  const float* pjb  = (const float*)d_in[8];
  const float* n2g  = (const float*)d_in[9];
  const float* n2b  = (const float*)d_in[10];
  const float* w1   = (const float*)d_in[11];
  const float* b1   = (const float*)d_in[12];
  const float* w2   = (const float*)d_in[13];
  const float* b2   = (const float*)d_in[14];
  float* out = (float*)d_out;

  char* p = (char*)d_ws;
  auto alloc = [&](size_t elems) -> _Float16* {
    _Float16* r = (_Float16*)p; p += elems * sizeof(_Float16); return r;
  };
  _Float16* qw_h = alloc((size_t)384 * 128);
  _Float16* pw_h = alloc((size_t)128 * 128);
  _Float16* w1_h = alloc((size_t)512 * 128);
  _Float16* w2_h = alloc((size_t)128 * 512);
  _Float16* xw_h = alloc((size_t)NWIN * NPAD * CDIM);   // reused as attn-out
  _Float16* q_h  = alloc((size_t)NWIN * 4 * NPAD * 32); // reused as LN2 out
  _Float16* k_h  = alloc((size_t)NWIN * 4 * NPAD * 32);
  _Float16* vt_h = alloc((size_t)NWIN * 4 * 32 * NPAD);
  _Float16* h1_h = alloc((size_t)NTOKENS * 512);
  _Float16* ao_h = xw_h;  // xw dead after k_qkv
  _Float16* m_h  = q_h;   // q dead after k_attn (8.39M >= 8.19M elems)

  k_cvt<<<192, 256, 0, stream>>>(qkvw, qw_h, 384 * 128);
  k_cvt<<<64,  256, 0, stream>>>(pjw,  pw_h, 128 * 128);
  k_cvt<<<256, 256, 0, stream>>>(w1,   w1_h, 512 * 128);
  k_cvt<<<256, 256, 0, stream>>>(w2,   w2_h, 128 * 512);

  k_ln1 <<<8192, 256, 0, stream>>>(x, n1g, n1b, xw_h);
  k_qkv <<<NWIN, 256, 0, stream>>>(xw_h, qw_h, qkvb, q_h, k_h, vt_h);
  k_attn<<<NWIN * 4, 256, 0, stream>>>(q_h, k_h, vt_h, relb, ao_h);
  k_proj<<<NWIN, 256, 0, stream>>>(ao_h, pw_h, pjb, x, out);
  k_ln2 <<<8000, 256, 0, stream>>>(out, n2g, n2b, m_h);
  k_mlp1<<<500, 256, 0, stream>>>(m_h, w1_h, b1, h1_h);
  k_mlp2<<<500, 256, 0, stream>>>(h1_h, w2_h, b2, out);
}
